// BandwidthActor_36550171689047
// MI455X (gfx1250) — matrix operand order repaired and checked by reading, not compiled, not run on hardware
//
#include <hip/hip_runtime.h>
#include <hip/hip_bf16.h>

// ---------------------------------------------------------------------------
// Types for CDNA5 WMMA
// ---------------------------------------------------------------------------
typedef __attribute__((ext_vector_type(16))) _Float16 v16h;
typedef __attribute__((ext_vector_type(8)))  float    v8f;

union FragH { v16h v; _Float16 h[16]; };

// ---------------------------------------------------------------------------
// Constants (from reference)
// ---------------------------------------------------------------------------
#define HID   128
#define HEADS 8
#define BW    10

__device__ __forceinline__ float gemm_act(float v, int act) {
  if (act == 1)      return v > 0.0f ? v : 0.0f;
  else if (act == 2) return 1.0f / (1.0f + __expf(-v));
  return v;
}

// ---------------------------------------------------------------------------
// WMMA GEMM, full-tile kernel: C[M,N] = act(A @ B + bias), M % 64 == 0.
// One wave computes a 64x16 tile: one (strided, scalar-load) B fragment is
// reused by 4 A fragments / 4 WMMAs per K-step. Ragged N handled by clamping
// the load column (always in-bounds) and masking only the store, so the inner
// loop is completely branch-free and EXEC is all-ones at every WMMA.
// K % 32 == 0 required. act: 0 = none, 1 = relu, 2 = sigmoid
// ---------------------------------------------------------------------------
__global__ __launch_bounds__(256) void k_wmma_gemm4(
    const float* __restrict__ A, int lda,
    const float* __restrict__ B, int ldb,
    const float* __restrict__ bias,
    float* __restrict__ C, int ldc,
    int N, int K, int act, int tiles, int tn) {
  int wave = blockIdx.x * 8 + (threadIdx.x >> 5);
  if (wave >= tiles) return;                  // wave-uniform: EXEC stays full
  int tm = wave / tn;                         // 64-row block index
  int tc = wave % tn;
  int lane = threadIdx.x & 31;
  int half = lane >> 4;
  int r    = lane & 15;
  int rowbase = tm * 64;
  int col     = tc * 16 + r;
  int colc    = col < N ? col : N - 1;        // clamped: loads always in-bounds

  v8f acc0 = {}, acc1 = {}, acc2 = {}, acc3 = {};
  for (int k0 = 0; k0 < K; k0 += 32) {
    // B fragment (32x16 f16): lane = column, elems = K run of 16.
    FragH fb;
    const float* bcol = B + (long)(k0 + half * 16) * ldb + colc;
#pragma unroll
    for (int e = 0; e < 16; e++) fb.h[e] = (_Float16)bcol[(long)e * ldb];

    // 4 A fragments (16x32 f16); lane half selects K sub-bands (ISA 7.12.2).
    FragH fa0, fa1, fa2, fa3;
    const float* arow = A + (long)(rowbase + r) * lda + k0 + half * 16;
#pragma unroll
    for (int e = 0; e < 8; e++) {
      fa0.h[e]     = (_Float16)arow[e];
      fa0.h[e + 8] = (_Float16)arow[e + 8];
      fa1.h[e]     = (_Float16)arow[(long)16 * lda + e];
      fa1.h[e + 8] = (_Float16)arow[(long)16 * lda + e + 8];
      fa2.h[e]     = (_Float16)arow[(long)32 * lda + e];
      fa2.h[e + 8] = (_Float16)arow[(long)32 * lda + e + 8];
      fa3.h[e]     = (_Float16)arow[(long)48 * lda + e];
      fa3.h[e + 8] = (_Float16)arow[(long)48 * lda + e + 8];
    }
    acc0 = __builtin_amdgcn_wmma_f32_16x16x32_f16(false, fa0.v, false, fb.v,
                                                  (short)0, acc0, false, false);
    acc1 = __builtin_amdgcn_wmma_f32_16x16x32_f16(false, fa1.v, false, fb.v,
                                                  (short)0, acc1, false, false);
    acc2 = __builtin_amdgcn_wmma_f32_16x16x32_f16(false, fa2.v, false, fb.v,
                                                  (short)0, acc2, false, false);
    acc3 = __builtin_amdgcn_wmma_f32_16x16x32_f16(false, fa3.v, false, fb.v,
                                                  (short)0, acc3, false, false);
  }

  if (col < N) {
    float bv = bias ? bias[col] : 0.0f;
#pragma unroll
    for (int rr = 0; rr < 8; rr++) {          // C layout: M = rr + 8*half
      int rowg = rowbase + rr + 8 * half;
      C[(long)(rowg)      * ldc + col] = gemm_act(acc0[rr] + bv, act);
      C[(long)(rowg + 16) * ldc + col] = gemm_act(acc1[rr] + bv, act);
      C[(long)(rowg + 32) * ldc + col] = gemm_act(acc2[rr] + bv, act);
      C[(long)(rowg + 48) * ldc + col] = gemm_act(acc3[rr] + bv, act);
    }
  }
}

// Tail kernel: one 16x16 tile per wave, M % 16 == 0 (covers M%64 leftovers).
__global__ __launch_bounds__(256) void k_wmma_gemm1(
    const float* __restrict__ A, int lda,
    const float* __restrict__ B, int ldb,
    const float* __restrict__ bias,
    float* __restrict__ C, int ldc,
    int N, int K, int act, int tiles, int tn) {
  int wave = blockIdx.x * 8 + (threadIdx.x >> 5);
  if (wave >= tiles) return;
  int tm = wave / tn;
  int tc = wave % tn;
  int lane = threadIdx.x & 31;
  int half = lane >> 4;
  int r    = lane & 15;
  int row0 = tm * 16;
  int col  = tc * 16 + r;
  int colc = col < N ? col : N - 1;

  v8f acc = {};
  for (int k0 = 0; k0 < K; k0 += 32) {
    FragH fa, fb;
    const float* bcol = B + (long)(k0 + half * 16) * ldb + colc;
#pragma unroll
    for (int e = 0; e < 16; e++) fb.h[e] = (_Float16)bcol[(long)e * ldb];
    const float* arow = A + (long)(row0 + r) * lda + k0 + half * 16;
#pragma unroll
    for (int e = 0; e < 8; e++) {
      fa.h[e]     = (_Float16)arow[e];
      fa.h[e + 8] = (_Float16)arow[e + 8];
    }
    acc = __builtin_amdgcn_wmma_f32_16x16x32_f16(false, fa.v, false, fb.v,
                                                 (short)0, acc, false, false);
  }
  if (col < N) {
    float bv = bias ? bias[col] : 0.0f;
#pragma unroll
    for (int rr = 0; rr < 8; rr++) {
      int rowg = row0 + rr + 8 * half;
      C[(long)rowg * ldc + col] = gemm_act(acc[rr] + bv, act);
    }
  }
}

static void run_gemm(const float* A, int lda, const float* B, int ldb,
                     const float* bias, float* C, int ldc,
                     int M, int N, int K, int act, hipStream_t s) {
  int tn = (N + 15) / 16;
  int Mfull = M & ~63;
  if (Mfull > 0) {
    int tiles = (Mfull / 64) * tn;
    k_wmma_gemm4<<<(tiles + 7) / 8, 256, 0, s>>>(A, lda, B, ldb, bias, C, ldc,
                                                 N, K, act, tiles, tn);
  }
  int Mtail = M - Mfull;                       // multiple of 16 (possibly 0)
  if (Mtail > 0) {
    int tiles = (Mtail / 16) * tn;
    k_wmma_gemm1<<<(tiles + 7) / 8, 256, 0, s>>>(
        A + (long)Mfull * lda, lda, B, ldb, bias, C + (long)Mfull * ldc, ldc,
        N, K, act, tiles, tn);
  }
}

// ---------------------------------------------------------------------------
// Small helper kernels
// ---------------------------------------------------------------------------
__global__ void k_fill(float* p, float v, int n) {
  int i = blockIdx.x * 256 + threadIdx.x;
  if (i < n) p[i] = v;
}

__device__ __forceinline__ void atomicMaxF(float* addr, float v) {
  if (v >= 0.0f) atomicMax((int*)addr, __float_as_int(v));
  else           atomicMin((unsigned int*)addr, (unsigned int)__float_as_int(v));
}

// a_src[n,h] = <h_row, att_src[h]>, a_dst likewise
__global__ void k_node_att(const float* __restrict__ h,
                           const float* __restrict__ att_s,
                           const float* __restrict__ att_d,
                           float* __restrict__ asrc, float* __restrict__ adst,
                           int Nn) {
  int t = blockIdx.x * 256 + threadIdx.x;
  if (t >= Nn * HEADS) return;
  int n = t >> 3, hh = t & 7;
  const float* hr = h + (long)n * (HEADS * HID) + hh * HID;
  const float* sp = att_s + hh * HID;
  const float* dp = att_d + hh * HID;
  float ss = 0.0f, dd = 0.0f;
  for (int c = 0; c < HID; c++) { float hv = hr[c]; ss += hv * sp[c]; dd += hv * dp[c]; }
  asrc[t] = ss;
  adst[t] = dd;
}

// segment max of leaky_relu(a_src[src]+a_dst[dst]) over dst
__global__ void k_edge_max(const int* __restrict__ ei, int E, int Nn,
                           const float* __restrict__ asrc,
                           const float* __restrict__ adst,
                           float* __restrict__ m) {
  int t = blockIdx.x * 256 + threadIdx.x;
  int ET = E + Nn;
  if (t >= ET * HEADS) return;
  int e = t >> 3, hh = t & 7;
  int s_, d_;
  if (e < E) { s_ = ei[e]; d_ = ei[E + e]; } else { s_ = d_ = e - E; }
  float v = asrc[s_ * HEADS + hh] + adst[d_ * HEADS + hh];
  v = v > 0.0f ? v : 0.2f * v;
  atomicMaxF(&m[d_ * HEADS + hh], v);
}

// e_exp = exp(e - m[dst]); denom[dst] += e_exp
__global__ void k_edge_exp(const int* __restrict__ ei, int E, int Nn,
                           const float* __restrict__ asrc,
                           const float* __restrict__ adst,
                           const float* __restrict__ m,
                           float* __restrict__ eexp,
                           float* __restrict__ denom) {
  int t = blockIdx.x * 256 + threadIdx.x;
  int ET = E + Nn;
  if (t >= ET * HEADS) return;
  int e = t >> 3, hh = t & 7;
  int s_, d_;
  if (e < E) { s_ = ei[e]; d_ = ei[E + e]; } else { s_ = d_ = e - E; }
  float v = asrc[s_ * HEADS + hh] + adst[d_ * HEADS + hh];
  v = v > 0.0f ? v : 0.2f * v;
  float ee = __expf(v - m[d_ * HEADS + hh]);
  eexp[t] = ee;
  atomicAdd(&denom[d_ * HEADS + hh], ee);
}

// out[dst,c] += (1/H) * sum_h alpha[e,h] * h[src, h*128 + c]   (block per edge)
__global__ void k_edge_agg(const int* __restrict__ ei, int E, int Nn,
                           const float* __restrict__ h,
                           const float* __restrict__ eexp,
                           const float* __restrict__ denom,
                           float* __restrict__ ob) {
  int e = blockIdx.x;
  int c = threadIdx.x;            // 128 threads
  int s_, d_;
  if (e < E) { s_ = ei[e]; d_ = ei[E + e]; } else { s_ = d_ = e - E; }
  __shared__ float al[HEADS];
  if (c < HEADS) al[c] = eexp[e * HEADS + c] / denom[d_ * HEADS + c];
  __syncthreads();
  const float* hr = h + (long)s_ * (HEADS * HID);
  float acc = 0.0f;
#pragma unroll
  for (int hh = 0; hh < HEADS; hh++) acc += al[hh] * hr[hh * HID + c];
  atomicAdd(&ob[(long)d_ * HID + c], acc * 0.125f);
}

__global__ void k_bias_relu(const float* __restrict__ ob,
                            const float* __restrict__ bias,
                            float* __restrict__ xb, int n) {
  int i = blockIdx.x * 256 + threadIdx.x;
  if (i < n) {
    float v = ob[i] + bias[i & (HID - 1)];
    xb[i] = v > 0.0f ? v : 0.0f;
  }
}

__global__ void k_colmean(const float* __restrict__ x, int rows,
                          float* __restrict__ mean) {
  int c = threadIdx.x;            // 128 threads, grid 1
  float s = 0.0f;
  for (int r = 0; r < rows; r++) s += x[(long)r * HID + c];
  mean[c] = s / (float)rows;
}

__global__ void k_make_links(int n, int L, int* __restrict__ ii,
                             int* __restrict__ jj, float* __restrict__ oidx) {
  int l = blockIdx.x * 256 + threadIdx.x;
  if (l >= L) return;
  int rem = l, i = 0;
  while (rem >= n - 1 - i) { rem -= n - 1 - i; i++; }
  int j = i + 1 + rem;
  ii[l] = i; jj[l] = j;
  oidx[2 * l]     = (float)i;
  oidx[2 * l + 1] = (float)j;
}

__global__ void k_gather_concat(const float* __restrict__ vf,
                                const int* __restrict__ ii,
                                const int* __restrict__ jj,
                                float* __restrict__ lf, int L) {
  int t = blockIdx.x * 256 + threadIdx.x;
  if (t >= L * 256) return;
  int l = t >> 8, c = t & 255;
  int node = (c < HID) ? ii[l] : jj[l];
  lf[t] = vf[node * HID + (c & (HID - 1))];
}

__global__ void k_gather_rows(const float* __restrict__ pe,
                              const int* __restrict__ map,
                              float* __restrict__ mp, int n) {
  int t = blockIdx.x * 256 + threadIdx.x;
  if (t >= n * HID) return;
  int i = t >> 7, c = t & (HID - 1);
  mp[t] = pe[(long)map[i] * HID + c];
}

// 8-head attention over 256 keys; also emits mean-over-heads weights to d_out.
__global__ __launch_bounds__(256) void k_attention(
    const float* __restrict__ Q, const float* __restrict__ Km,
    const float* __restrict__ V, float* __restrict__ attn,
    float* __restrict__ wout, int L) {
  int l = blockIdx.x;
  if (l >= L) return;
  int t = threadIdx.x;
  __shared__ float qs[HID];
  __shared__ float sc[HEADS * 256];
  __shared__ float mx[HEADS], sm[HEADS];
  if (t < HID) qs[t] = Q[(long)l * HID + t];
  __syncthreads();
  {
    const float* kr = Km + (long)t * HID;
#pragma unroll
    for (int hh = 0; hh < HEADS; hh++) {
      float a = 0.0f;
#pragma unroll
      for (int d = 0; d < 16; d++) a += qs[hh * 16 + d] * kr[hh * 16 + d];
      sc[hh * 256 + t] = a * 0.25f;           // 1/sqrt(16)
    }
  }
  __syncthreads();
  if (t < HEADS) {
    float m = -3.0e38f;
    for (int n = 0; n < 256; n++) { float v = sc[t * 256 + n]; m = v > m ? v : m; }
    float s = 0.0f;
    for (int n = 0; n < 256; n++) s += __expf(sc[t * 256 + n] - m);
    mx[t] = m; sm[t] = s;
  }
  __syncthreads();
  {
    float wm = 0.0f;
#pragma unroll
    for (int hh = 0; hh < HEADS; hh++) {
      float w = __expf(sc[hh * 256 + t] - mx[hh]) / sm[hh];
      sc[hh * 256 + t] = w;
      wm += w;
    }
    wout[(long)l * 256 + t] = wm * 0.125f;
  }
  __syncthreads();
  if (t < HID) {
    int hh = t >> 4;
    float a = 0.0f;
    for (int n = 0; n < 256; n++) a += sc[hh * 256 + n] * V[(long)n * HID + t];
    attn[(long)l * HID + t] = a;
  }
}

// effective bias: b[j] + sum_k vm[k]*W[128+k,j] + sum_k pm[k]*W[256+k,j]
__global__ void k_eff_bias(const float* __restrict__ W, int ldb,
                           const float* __restrict__ b,
                           const float* __restrict__ vm,
                           const float* __restrict__ pm,
                           float* __restrict__ out, int N) {
  int j = blockIdx.x * 256 + threadIdx.x;
  if (j >= N) return;
  float acc = b[j];
  for (int k = 0; k < HID; k++) acc += vm[k] * W[(long)(HID + k) * ldb + j];
  for (int k = 0; k < HID; k++) acc += pm[k] * W[(long)(2 * HID + k) * ldb + j];
  out[j] = acc;
}

__global__ void k_broadcast_cs(const float* __restrict__ cs,
                               float* __restrict__ out, int L) {
  int t = blockIdx.x * 256 + threadIdx.x;
  if (t < L * BW) out[t] = cs[t / BW];
}

// ---------------------------------------------------------------------------
// Host-side encoder driver
// ---------------------------------------------------------------------------
static void run_encoder(const float* x0, const int* ei, int E, int Nn,
                        const float* const W[3], const float* const as_[3],
                        const float* const ad_[3], const float* const bi[3],
                        const float* projW, const float* projB, float* pe_out,
                        float* h, float* xb, float* ob,
                        float* asrc, float* adst, float* m, float* denom,
                        float* eexp, hipStream_t s) {
  int ET = E + Nn;
  const float* xin = x0;
  for (int layer = 0; layer < 3; layer++) {
    run_gemm(xin, HID, W[layer], HEADS * HID, nullptr, h, HEADS * HID,
             Nn, HEADS * HID, HID, 0, s);
    k_node_att<<<(Nn * HEADS + 255) / 256, 256, 0, s>>>(h, as_[layer], ad_[layer],
                                                        asrc, adst, Nn);
    k_fill<<<(Nn * HEADS + 255) / 256, 256, 0, s>>>(m, -3.0e38f, Nn * HEADS);
    k_fill<<<(Nn * HEADS + 255) / 256, 256, 0, s>>>(denom, 0.0f, Nn * HEADS);
    k_fill<<<(Nn * HID + 255) / 256, 256, 0, s>>>(ob, 0.0f, Nn * HID);
    k_edge_max<<<(ET * HEADS + 255) / 256, 256, 0, s>>>(ei, E, Nn, asrc, adst, m);
    k_edge_exp<<<(ET * HEADS + 255) / 256, 256, 0, s>>>(ei, E, Nn, asrc, adst, m,
                                                        eexp, denom);
    k_edge_agg<<<ET, HID, 0, s>>>(ei, E, Nn, h, eexp, denom, ob);
    k_bias_relu<<<(Nn * HID + 255) / 256, 256, 0, s>>>(ob, bi[layer], xb, Nn * HID);
    xin = xb;
  }
  run_gemm(xb, HID, projW, HID, projB, pe_out, HID, Nn, HID, HID, 0, s);
}

// ---------------------------------------------------------------------------
// Entry point
// ---------------------------------------------------------------------------
extern "C" void kernel_launch(void* const* d_in, const int* in_sizes, int n_in,
                              void* d_out, int out_size, void* d_ws, size_t ws_size,
                              hipStream_t stream) {
  (void)n_in; (void)out_size; (void)ws_size;
  const float* phys_feat = (const float*)d_in[0];
  const int*   p_ei      = (const int*)d_in[1];
  const float* virt_feat = (const float*)d_in[3];
  const int*   v_ei      = (const int*)d_in[4];
  const int*   mapping   = (const int*)d_in[6];

  int NP = in_sizes[0] / HID;       // 10000
  int EP = in_sizes[1] / 2;         // 160000
  int NV = in_sizes[3] / HID;       // 256
  int EV = in_sizes[4] / 2;         // 4096
  int L  = NV * (NV - 1) / 2;       // 32640

  auto F = [&](int i) { return (const float*)d_in[i]; };
  int pi = 7;
  // physical encoder params
  const float *pW[3], *pAS[3], *pAD[3], *pBI[3];
  for (int c = 0; c < 3; c++) { pW[c] = F(pi++); pAS[c] = F(pi++); pAD[c] = F(pi++); pBI[c] = F(pi++); }
  const float* p_projW = F(pi++); const float* p_projB = F(pi++);
  // virtual encoder params
  const float *vW[3], *vAS[3], *vAD[3], *vBI[3];
  for (int c = 0; c < 3; c++) { vW[c] = F(pi++); vAS[c] = F(pi++); vAD[c] = F(pi++); vBI[c] = F(pi++); }
  const float* v_projW = F(pi++); const float* v_projB = F(pi++);
  const float* link_W1 = F(pi++); const float* link_b1 = F(pi++);
  const float* link_W2 = F(pi++); const float* link_b2 = F(pi++);
  const float* att_Wq = F(pi++); const float* att_bq = F(pi++);
  const float* att_Wk = F(pi++); const float* att_bk = F(pi++);
  const float* att_Wv = F(pi++); const float* att_bv = F(pi++);
  const float* att_Wo = F(pi++); const float* att_bo = F(pi++);
  const float* gb_W1 = F(pi++); const float* gb_b1 = F(pi++);
  const float* gb_W2 = F(pi++); const float* gb_b2 = F(pi++);
  const float* gb_W3 = F(pi++); const float* gb_b3 = F(pi++);
  const float* cc_W1 = F(pi++); const float* cc_b1 = F(pi++);
  const float* cc_W2 = F(pi++); const float* cc_b2 = F(pi++);
  const float* cc_W3 = F(pi++); const float* cc_b3 = F(pi++);

  // ---------------- workspace layout (static, region-reused) ----------------
  char* ws = (char*)d_ws;
  const size_t OFF_A = 0;                       // 40,960,000 : h_buf -> h1
  const size_t OFF_B = 40960000;                //  5,120,000 : x_buf -> small bufs
  const size_t OFF_C = 46080000;                //  8,355,840 : out_buf -> c2
  const size_t OFF_D = 54435840;                //  1,280,000 : a/m/denom -> ii/jj
  const size_t OFF_E = 55715840;                //  5,440,000 : e_exp -> cs
  const size_t OFF_F = 61155840;                //  5,120,000 : pe (persist)
  const size_t OFF_G = 66275840;                //    131,072 : ve (persist)
  const size_t OFF_H = 66406912;                // 33,423,360 : link_feat -> attn|attended
  const size_t OFF_I = 99830272;                // 16,711,680 : le -> c1
  const size_t OFF_J = 116541952;               // 16,711,680 : t1 -> q -> h2

  float* h_buf = (float*)(ws + OFF_A);
  float* x_buf = (float*)(ws + OFF_B);
  float* o_buf = (float*)(ws + OFF_C);
  float* asrc  = (float*)(ws + OFF_D);
  float* adst  = (float*)(ws + OFF_D + 320000);
  float* mbuf  = (float*)(ws + OFF_D + 640000);
  float* dbuf  = (float*)(ws + OFF_D + 960000);
  float* eexp  = (float*)(ws + OFF_E);
  float* pe    = (float*)(ws + OFF_F);
  float* ve    = (float*)(ws + OFF_G);

  // ---------------- encoders ----------------
  run_encoder(phys_feat, p_ei, EP, NP, pW, pAS, pAD, pBI, p_projW, p_projB, pe,
              h_buf, x_buf, o_buf, asrc, adst, mbuf, dbuf, eexp, stream);
  run_encoder(virt_feat, v_ei, EV, NV, vW, vAS, vAD, vBI, v_projW, v_projB, ve,
              h_buf, x_buf, o_buf, asrc, adst, mbuf, dbuf, eexp, stream);

  // small buffers carved from region B (x_buf dead now)
  float* vm      = (float*)(ws + OFF_B);
  float* pm      = (float*)(ws + OFF_B + 512);
  float* mp      = (float*)(ws + OFF_B + 1024);
  float* kk      = (float*)(ws + OFF_B + 1024 + 131072);
  float* vv      = (float*)(ws + OFF_B + 1024 + 2 * 131072);
  float* bias_gb = (float*)(ws + OFF_B + 1024 + 3 * 131072);
  float* bias_cc = (float*)(ws + OFF_B + 1024 + 3 * 131072 + 1024);
  int*   ii      = (int*)(ws + OFF_D);
  int*   jj      = (int*)(ws + OFF_D + 4 * (size_t)L);

  k_colmean<<<1, HID, 0, stream>>>(pe, NP, pm);
  k_colmean<<<1, HID, 0, stream>>>(ve, NV, vm);

  // ---------------- link construction ----------------
  float* out_logits = (float*)d_out;
  float* out_cs     = out_logits + (size_t)L * BW;
  float* out_w      = out_logits + 2 * (size_t)L * BW;
  float* out_idx    = out_w + (size_t)L * 256;

  k_make_links<<<(L + 255) / 256, 256, 0, stream>>>(NV, L, ii, jj, out_idx);

  float* link_feat = (float*)(ws + OFF_H);
  k_gather_concat<<<((size_t)L * 256 + 255) / 256, 256, 0, stream>>>(
      virt_feat, ii, jj, link_feat, L);

  float* t1 = (float*)(ws + OFF_J);
  float* le = (float*)(ws + OFF_I);
  run_gemm(link_feat, 2 * HID, link_W1, HID, link_b1, t1, HID, L, HID, 2 * HID, 1, stream);
  run_gemm(t1, HID, link_W2, HID, link_b2, le, HID, L, HID, HID, 0, stream);

  // ---------------- attention ----------------
  k_gather_rows<<<(NV * HID + 255) / 256, 256, 0, stream>>>(pe, mapping, mp, NV);

  float* q = (float*)(ws + OFF_J);              // t1 dead
  run_gemm(le, HID, att_Wq, HID, att_bq, q,  HID, L,  HID, HID, 0, stream);
  run_gemm(mp, HID, att_Wk, HID, att_bk, kk, HID, NV, HID, HID, 0, stream);
  run_gemm(mp, HID, att_Wv, HID, att_bv, vv, HID, NV, HID, HID, 0, stream);

  float* attn     = (float*)(ws + OFF_H);                 // link_feat dead
  float* attended = (float*)(ws + OFF_H + 16711680);
  k_attention<<<L, 256, 0, stream>>>(q, kk, vv, attn, out_w, L);
  run_gemm(attn, HID, att_Wo, HID, att_bo, attended, HID, L, HID, HID, 0, stream);

  // ---------------- heads (comb folded into effective biases) ----------------
  k_eff_bias<<<1, 256, 0, stream>>>(gb_W1, 2 * HID, gb_b1, vm, pm, bias_gb, 2 * HID);
  k_eff_bias<<<1, 256, 0, stream>>>(cc_W1, HID,     cc_b1, vm, pm, bias_cc, HID);

  float* h1 = (float*)(ws + OFF_A);             // h_buf dead
  float* h2 = (float*)(ws + OFF_J);             // q dead
  run_gemm(attended, HID, gb_W1, 2 * HID, bias_gb, h1, 2 * HID, L, 2 * HID, HID, 1, stream);
  run_gemm(h1, 2 * HID, gb_W2, HID, gb_b2, h2, HID, L, HID, 2 * HID, 1, stream);
  run_gemm(h2, HID, gb_W3, BW, gb_b3, out_logits, BW, L, BW, HID, 0, stream);

  float* c1 = (float*)(ws + OFF_I);             // le dead
  float* c2 = (float*)(ws + OFF_C);             // out_buf dead
  float* cs = (float*)(ws + OFF_E);             // e_exp dead
  run_gemm(attended, HID, cc_W1, HID, bias_cc, c1, HID, L, HID, HID, 1, stream);
  run_gemm(c1, HID, cc_W2, HID / 2, cc_b2, c2, HID / 2, L, HID / 2, HID, 1, stream);
  run_gemm(c2, HID / 2, cc_W3, 1, cc_b3, cs, 1, L, 1, HID / 2, 2, stream);
  k_broadcast_cs<<<(L * BW + 255) / 256, 256, 0, stream>>>(cs, out_cs, L);
}
